// MultiHeadAttention_82497731821806
// MI455X (gfx1250) — matrix operand order repaired and checked by reading, not compiled, not run on hardware
//
#include <hip/hip_runtime.h>
#include <stdint.h>

// ---------------------------------------------------------------------------
// MI455X / gfx1250 multi-head attention forward.
// BN=4, T=2048, D_MODEL=1024, H=16, DH=64.
// All matmuls via v_wmma_f32_16x16x32_bf16 (fp32 accumulate).
// Tail GEMM A-tile staged with global_load_async_to_lds_b128 (ASYNCcnt path).
// ---------------------------------------------------------------------------

typedef __bf16 bf16;
typedef bf16  v16bf __attribute__((ext_vector_type(16)));
typedef float v8f   __attribute__((ext_vector_type(8)));

#define DEV __device__ __forceinline__

#define T_SEQ    2048
#define DM       1024
#define NH       16
#define DH       64
#define BATCH    4
#define M_ROWS   (BATCH * T_SEQ)   // 8192

// GEMM tiling
#define BM  128
#define BNT 128
#define BK  32
#define KP  40   // padded LDS row stride in bf16 (80 B/row -> 16B-aligned rows)

DEV v8f v8f_zero() {
  v8f z;
#pragma unroll
  for (int i = 0; i < 8; ++i) z[i] = 0.0f;
  return z;
}

DEV v8f wmma_bf16(v16bf a, v16bf b, v8f c) {
  // 8 args: (neg_a, A, neg_b, B, c_mod, C, reuse_a, reuse_b)
  return __builtin_amdgcn_wmma_f32_16x16x32_bf16(
      false, a, false, b, (short)0, c, false, false);
}

// Async 16B copy global -> LDS (gfx1250 GLOBAL_LOAD_ASYNC_TO_LDS_B128).
// lds_addr: byte offset within the workgroup LDS allocation (low 32 bits of the
// flat pointer -- the LDS aperture keeps the offset in addr[31:0]).
DEV void async_load_b128(unsigned lds_addr, const void* gaddr) {
  asm volatile("global_load_async_to_lds_b128 %0, %1, off"
               :: "v"(lds_addr), "v"(gaddr)
               : "memory");
}

DEV void wait_async0() {
#if __has_builtin(__builtin_amdgcn_s_wait_asynccnt)
  __builtin_amdgcn_s_wait_asynccnt(0);
#else
  asm volatile("s_wait_asynccnt 0x0" ::: "memory");
#endif
}

// A fragment (16xK=32, bf16), ISA 7.12.2 layout:
//   lane&15 = row M; lanes<16: K=0..7,16..23 ; lanes>=16: K=8..15,24..31
//   2 bf16 packed per VGPR -> 32-bit reads.
DEV v16bf load_a_frag(const bf16* tile, int ld, int lane) {
  const int row = lane & 15;
  const int hi  = (lane >> 4) & 1;
  const unsigned* r = (const unsigned*)(tile + row * ld);
  union { v16bf v; unsigned u[8]; } f;
#pragma unroll
  for (int i = 0; i < 4; ++i) f.u[i]     = r[4 * hi + i];      // K = 8*hi + 2i
#pragma unroll
  for (int i = 0; i < 4; ++i) f.u[4 + i] = r[8 + 4 * hi + i];  // K = 16 + 8*hi + 2i
  return f.v;
}

// B fragment (K=32 x 16, bf16): lane&15 = col N; lanes<16: K=0..7,16..23 ; lanes>=16: K=8..15,24..31.
// Source is stored N-major with contiguous K per row (leading dim = ld).
DEV v16bf load_b_frag(const bf16* tile, int ld, int lane) {
  const int col = lane & 15;
  const int hi  = (lane >> 4) & 1;
  const unsigned* r = (const unsigned*)(tile + (size_t)col * ld);
  union { v16bf v; unsigned u[8]; } f;
#pragma unroll
  for (int i = 0; i < 4; ++i) f.u[i]     = r[4 * hi + i];      // K = 8*hi + 2i
#pragma unroll
  for (int i = 0; i < 4; ++i) f.u[4 + i] = r[8 + 4 * hi + i];  // K = 16 + 8*hi + 2i
  return f.v;
}

// ---------------------------------------------------------------------------
// Kernel 1: qkv = x @ w_qkv^T ; scatter into Q[b,h,t,d], K[b,h,t,d], V^T[b,h,d,t]
// (fp32 sources -> convert to bf16 while staging to LDS; async copy not usable
//  here because the DMA cannot down-convert.)
// ---------------------------------------------------------------------------
__global__ __launch_bounds__(256) void qkv_gemm_kernel(
    const float* __restrict__ x, const float* __restrict__ w,
    bf16* __restrict__ Qb, bf16* __restrict__ Kb, bf16* __restrict__ Vt) {
  __shared__ bf16 As[BM * KP];
  __shared__ bf16 Bs[BNT * KP];
  const int tid = threadIdx.x, lane = tid & 31, wid = tid >> 5;
  const int n0 = blockIdx.x * BNT, m0 = blockIdx.y * BM;
  const int wm = (wid & 3) * 32, wn = (wid >> 2) * 64;

  v8f acc[2][4];
#pragma unroll
  for (int i = 0; i < 2; ++i)
#pragma unroll
    for (int j = 0; j < 4; ++j) acc[i][j] = v8f_zero();

  const int srow = tid >> 1, scol = (tid & 1) * 16;
  const float* ag = x + (size_t)(m0 + srow) * DM + scol;
  const float* bg = w + (size_t)(n0 + srow) * DM + scol;
  bf16* asr = As + srow * KP + scol;
  bf16* bsr = Bs + srow * KP + scol;

  for (int k0 = 0; k0 < DM; k0 += BK) {
    if (k0 + BK < DM) {  // gfx1250 global_prefetch of next K tile
      __builtin_prefetch(ag + k0 + BK, 0, 1);
      __builtin_prefetch(bg + k0 + BK, 0, 1);
    }
#pragma unroll
    for (int i = 0; i < 16; i += 4) {
      float4 va = *(const float4*)(ag + k0 + i);
      float4 vb = *(const float4*)(bg + k0 + i);
      asr[i + 0] = (bf16)va.x; asr[i + 1] = (bf16)va.y;
      asr[i + 2] = (bf16)va.z; asr[i + 3] = (bf16)va.w;
      bsr[i + 0] = (bf16)vb.x; bsr[i + 1] = (bf16)vb.y;
      bsr[i + 2] = (bf16)vb.z; bsr[i + 3] = (bf16)vb.w;
    }
    __syncthreads();
    v16bf aF[2], bF[4];
    aF[0] = load_a_frag(As + (wm + 0) * KP, KP, lane);
    aF[1] = load_a_frag(As + (wm + 16) * KP, KP, lane);
#pragma unroll
    for (int j = 0; j < 4; ++j)
      bF[j] = load_b_frag(Bs + (wn + j * 16) * KP, KP, lane);
#pragma unroll
    for (int i = 0; i < 2; ++i)
#pragma unroll
      for (int j = 0; j < 4; ++j) acc[i][j] = wmma_bf16(aF[i], bF[j], acc[i][j]);
    __syncthreads();
  }

  // C layout: element e of v8f = row M = e + 8*(lane>>4), col N = lane&15
  const int hi = (lane >> 4) & 1, col = lane & 15;
#pragma unroll
  for (int i = 0; i < 2; ++i) {
    const int mbase = m0 + wm + i * 16 + 8 * hi;
#pragma unroll
    for (int j = 0; j < 4; ++j) {
      const int gn = n0 + wn + j * 16 + col;      // output feature: h*192 + c*64 + d
      const int h = gn / 192, rem = gn % 192;
      const int c = rem >> 6, d = rem & 63;
#pragma unroll
      for (int e = 0; e < 8; ++e) {
        const int gm = mbase + e;
        const int bn = gm >> 11, t = gm & (T_SEQ - 1);
        const size_t bh = (size_t)(bn * NH + h);
        const bf16 val = (bf16)acc[i][j][e];
        if (c == 0)      Qb[(bh * T_SEQ + t) * DH + d] = val;
        else if (c == 1) Kb[(bh * T_SEQ + t) * DH + d] = val;
        else             Vt[(bh * DH + d) * T_SEQ + t] = val;   // transposed V
      }
    }
  }
}

// ---------------------------------------------------------------------------
// Kernel 2: flash attention. One block = (bn, h, 128 queries); 8 waves,
// each wave owns a 16-query tile. Online softmax over 32-key tiles.
// ---------------------------------------------------------------------------
__global__ __launch_bounds__(256) void attn_kernel(
    const bf16* __restrict__ Qb, const bf16* __restrict__ Kb,
    const bf16* __restrict__ Vt, const int* __restrict__ mask,
    bf16* __restrict__ Ab) {
  __shared__ bf16 Plds[8][16 * 36];  // wave-private 16x32 P tile (padded)
  const int tid = threadIdx.x, lane = tid & 31, wid = tid >> 5;
  const int col = lane & 15, hi = (lane >> 4) & 1;
  const int blk = blockIdx.x;
  const int qb = blk & 15, h = (blk >> 4) & 15, bn = blk >> 8;
  const int q0 = qb * 128 + wid * 16;
  const size_t bh = (size_t)(bn * NH + h);
  const bf16* Qt = Qb + (bh * T_SEQ + q0) * DH;
  const bf16* Kt = Kb + bh * T_SEQ * DH;
  const bf16* Vb = Vt + bh * DH * T_SEQ;
  const int*  mr = mask + bn * T_SEQ;

  // Q tile held in registers as two A fragments (K = dh halves)
  const v16bf qA0 = load_a_frag(Qt, DH, lane);
  const v16bf qA1 = load_a_frag(Qt + 32, DH, lane);

  float mrun[8], lrun[8];
  v8f oacc[4];
#pragma unroll
  for (int e = 0; e < 8; ++e) { mrun[e] = -1e30f; lrun[e] = 0.0f; }
#pragma unroll
  for (int n = 0; n < 4; ++n) oacc[n] = v8f_zero();

  bf16* Pw = &Plds[wid][0];
  const float scale = 0.125f;  // DH^-0.5

  for (int kt = 0; kt < T_SEQ; kt += 32) {
    v8f s0 = v8f_zero(), s1 = v8f_zero();
    const bf16* Kt0 = Kt + (size_t)kt * DH;
    v16bf kb;
    kb = load_b_frag(Kt0,                 DH, lane); s0 = wmma_bf16(qA0, kb, s0);
    kb = load_b_frag(Kt0 + 32,            DH, lane); s0 = wmma_bf16(qA1, kb, s0);
    kb = load_b_frag(Kt0 + 16 * DH,       DH, lane); s1 = wmma_bf16(qA0, kb, s1);
    kb = load_b_frag(Kt0 + 16 * DH + 32,  DH, lane); s1 = wmma_bf16(qA1, kb, s1);

    // lane's whole accumulator column shares one key -> one mask bit per group
    const bool ok0 = mr[kt + col] != 0;
    const bool ok1 = mr[kt + 16 + col] != 0;
#pragma unroll
    for (int e = 0; e < 8; ++e) {
      float a = ok0 ? s0[e] * scale : -1e30f;
      float b = ok1 ? s1[e] * scale : -1e30f;
      float mx = fmaxf(a, b);
#pragma unroll
      for (int off = 1; off < 16; off <<= 1)
        mx = fmaxf(mx, __shfl_xor(mx, off, 32));   // row-max across 16 lanes/half
      const float mn    = fmaxf(mrun[e], mx);
      const float alpha = __expf(mrun[e] - mn);
      const float pa = __expf(a - mn);
      const float pb = __expf(b - mn);
      float sum = pa + pb;
#pragma unroll
      for (int off = 1; off < 16; off <<= 1)
        sum += __shfl_xor(sum, off, 32);           // row-sum across 16 lanes/half
      lrun[e] = lrun[e] * alpha + sum;
      mrun[e] = mn;
#pragma unroll
      for (int n = 0; n < 4; ++n) oacc[n][e] *= alpha;
      // relayout P from C layout to A layout via wave-private LDS
      Pw[(e + 8 * hi) * 36 + col]      = (bf16)pa;
      Pw[(e + 8 * hi) * 36 + 16 + col] = (bf16)pb;
    }
    asm volatile("s_wait_dscnt 0x0" ::: "memory");  // in-wave LDS RAW fence
    const v16bf pA = load_a_frag(Pw, 36, lane);
#pragma unroll
    for (int n = 0; n < 4; ++n) {
      const v16bf vB = load_b_frag(Vb + (size_t)(n * 16) * T_SEQ + kt, T_SEQ, lane);
      oacc[n] = wmma_bf16(pA, vB, oacc[n]);
    }
  }

#pragma unroll
  for (int e = 0; e < 8; ++e) {
    const float rinv = 1.0f / lrun[e];
    const int q = q0 + e + 8 * hi;
#pragma unroll
    for (int n = 0; n < 4; ++n) {
      Ab[((size_t)bn * T_SEQ + q) * DM + h * DH + n * 16 + col] =
          (bf16)(oacc[n][e] * rinv);
    }
  }
}

// ---------------------------------------------------------------------------
// Kernel 3: out = attn_flat @ w_tail^T + b_tail   (fp32 output)
// A tile is already bf16 -> stage it with async DMA to LDS (no VGPR round trip).
// ---------------------------------------------------------------------------
__global__ __launch_bounds__(256) void tail_gemm_kernel(
    const bf16* __restrict__ A, const float* __restrict__ w,
    const float* __restrict__ bias, float* __restrict__ out) {
  __shared__ bf16 As[BM * KP];
  __shared__ bf16 Bs[BNT * KP];
  const int tid = threadIdx.x, lane = tid & 31, wid = tid >> 5;
  const int n0 = blockIdx.x * BNT, m0 = blockIdx.y * BM;
  const int wm = (wid & 3) * 32, wn = (wid >> 2) * 64;

  v8f acc[2][4];
#pragma unroll
  for (int i = 0; i < 2; ++i)
#pragma unroll
    for (int j = 0; j < 4; ++j) acc[i][j] = v8f_zero();

  // B staging (fp32 -> bf16 conversion in registers)
  const int srow = tid >> 1, scol = (tid & 1) * 16;
  const float* bg = w + (size_t)(n0 + srow) * DM + scol;
  bf16* bsr = Bs + srow * KP + scol;

  // A staging via async DMA: 128 rows x 32 bf16 = 512 x 16B chunks, 2 per thread
  const int ar0 = tid >> 2,         aq0 = (tid & 3) * 8;         // chunk 0
  const int ar1 = (tid + 256) >> 2, aq1 = ((tid + 256) & 3) * 8; // chunk 1
  const bf16* ag0 = A + (size_t)(m0 + ar0) * DM + aq0;
  const bf16* ag1 = A + (size_t)(m0 + ar1) * DM + aq1;
  const unsigned ad0 = (unsigned)(uintptr_t)(As + ar0 * KP + aq0);
  const unsigned ad1 = (unsigned)(uintptr_t)(As + ar1 * KP + aq1);

  for (int k0 = 0; k0 < DM; k0 += BK) {
    if (k0 + BK < DM)
      __builtin_prefetch(bg + k0 + BK, 0, 1);

    async_load_b128(ad0, ag0 + k0);   // DMA global -> LDS, no VGPR data
    async_load_b128(ad1, ag1 + k0);

#pragma unroll
    for (int i = 0; i < 16; i += 4) {
      float4 vb = *(const float4*)(bg + k0 + i);
      bsr[i + 0] = (bf16)vb.x; bsr[i + 1] = (bf16)vb.y;
      bsr[i + 2] = (bf16)vb.z; bsr[i + 3] = (bf16)vb.w;
    }
    wait_async0();                    // s_wait_asynccnt 0
    __syncthreads();
    v16bf aF[2], bF[4];
    aF[0] = load_a_frag(As + (wm + 0) * KP, KP, lane);
    aF[1] = load_a_frag(As + (wm + 16) * KP, KP, lane);
#pragma unroll
    for (int j = 0; j < 4; ++j)
      bF[j] = load_b_frag(Bs + (wn + j * 16) * KP, KP, lane);
#pragma unroll
    for (int i = 0; i < 2; ++i)
#pragma unroll
      for (int j = 0; j < 4; ++j) acc[i][j] = wmma_bf16(aF[i], bF[j], acc[i][j]);
    __syncthreads();
  }

  const int hi = (lane >> 4) & 1, col = lane & 15;
#pragma unroll
  for (int i = 0; i < 2; ++i) {
    const int mbase = m0 + wm + i * 16 + 8 * hi;
#pragma unroll
    for (int j = 0; j < 4; ++j) {
      const int gn = n0 + wn + j * 16 + col;
      const float bv = bias[gn];
#pragma unroll
      for (int e = 0; e < 8; ++e)
        out[(size_t)(mbase + e) * DM + gn] = acc[i][j][e] + bv;
    }
  }
}

// ---------------------------------------------------------------------------
extern "C" void kernel_launch(void* const* d_in, const int* in_sizes, int n_in,
                              void* d_out, int out_size, void* d_ws, size_t ws_size,
                              hipStream_t stream) {
  const float* x      = (const float*)d_in[0];
  const int*   mask   = (const int*)d_in[1];
  const float* w_qkv  = (const float*)d_in[2];
  const float* w_tail = (const float*)d_in[3];
  const float* b_tail = (const float*)d_in[4];
  float* out = (float*)d_out;

  const size_t QKV_ELEMS = (size_t)BATCH * NH * T_SEQ * DH;  // 8,388,608 bf16 each
  bf16* Qb = (bf16*)d_ws;
  bf16* Kb = Qb + QKV_ELEMS;
  bf16* Vt = Kb + QKV_ELEMS;
  bf16* Ab = Vt + QKV_ELEMS;  // [BN, T, H*DH] bf16  (total ws: 64 MiB)

  qkv_gemm_kernel<<<dim3(3 * DM / BNT, M_ROWS / BM), 256, 0, stream>>>(
      x, w_qkv, Qb, Kb, Vt);
  attn_kernel<<<dim3(BATCH * NH * (T_SEQ / 128)), 256, 0, stream>>>(
      Qb, Kb, Vt, mask, Ab);
  tail_gemm_kernel<<<dim3(DM / BNT, M_ROWS / BM), 256, 0, stream>>>(
      Ab, w_tail, b_tail, out);
}
